// SelectiveDiagonalSSM_58591943852307
// MI455X (gfx1250) — matrix operand order repaired and checked by reading, not compiled, not run on hardware
//
#include <hip/hip_runtime.h>

// Problem constants (match reference)
#define BATCH 4
#define TLEN  2048
#define HDIM  1024
#define NST   16
#define NCHUNK 16
#define CLEN  (TLEN / NCHUNK)   // 128 steps per chunk

typedef __attribute__((ext_vector_type(16))) __bf16 v16bf;
typedef __attribute__((ext_vector_type(8)))  __bf16 v8bf;
typedef __attribute__((ext_vector_type(8)))  float  v8f;

// ---------------------------------------------------------------------------
// Kernel 0: convert x (B*T*H f32) and W_dt (H*H f32) to bf16 in workspace.
// ---------------------------------------------------------------------------
__global__ __launch_bounds__(256) void convert_bf16_kernel(
    const float* __restrict__ x, const float* __restrict__ w,
    __bf16* __restrict__ xb, __bf16* __restrict__ wb, int nx, int nw) {
  int stride = gridDim.x * blockDim.x;
  for (int i = blockIdx.x * blockDim.x + threadIdx.x; i < nx; i += stride)
    xb[i] = (__bf16)x[i];
  for (int i = blockIdx.x * blockDim.x + threadIdx.x; i < nw; i += stride)
    wb[i] = (__bf16)w[i];
}

__device__ __forceinline__ float softplusf(float v) {
  // stable: max(v,0) + log1p(exp(-|v|))
  return fmaxf(v, 0.0f) + __logf(1.0f + __expf(-fabsf(v)));
}

// ---------------------------------------------------------------------------
// Kernel 1: dt GEMM.  dt[m,g] = softplus( sum_k u[m,k]*W[g,k] + b_dt[g] )
// Block = 128 threads (4 waves); wave tile 64x64 = 4x4 WMMA accumulators.
// A/B fragments loaded straight from global in the ISA VGPR layouts; both
// bf16 operands (18.9MB) are L2-resident (192MB L2) so no LDS staging.
// ---------------------------------------------------------------------------
__global__ __launch_bounds__(128) void dt_gemm_wmma_kernel(
    const __bf16* __restrict__ u,     // [M, HDIM] bf16
    const __bf16* __restrict__ w,     // [HDIM, HDIM] bf16 (row g, col h)
    const float*  __restrict__ b_dt,  // [HDIM]
    float*        __restrict__ dtbuf) // [M, HDIM] f32 out
{
  const int lane = threadIdx.x & 31;
  const int wave = threadIdx.x >> 5;
  const int m0 = blockIdx.x * 64;
  const int n0 = blockIdx.y * 256 + wave * 64;

  const int sel  = lane >> 4;
  const int lrow = lane & 15;
  const int kbA  = sel * 8;
  const int kbB  = sel * 8;

  v8f acc[4][4];
#pragma unroll
  for (int mi = 0; mi < 4; ++mi)
#pragma unroll
    for (int ni = 0; ni < 4; ++ni)
      acc[mi][ni] = (v8f)0.0f;

  for (int k0 = 0; k0 < HDIM; k0 += 32) {
    v16bf afrag[4], bfrag[4];
#pragma unroll
    for (int mi = 0; mi < 4; ++mi) {
      const __bf16* ap = u + (size_t)(m0 + mi * 16 + lrow) * HDIM + k0 + kbA;
      v8bf lo = *(const v8bf*)ap;          // K = kbA .. kbA+7
      v8bf hi = *(const v8bf*)(ap + 16);   // K = kbA+16 .. kbA+23
      afrag[mi] = __builtin_shufflevector(lo, hi, 0, 1, 2, 3, 4, 5, 6, 7,
                                          8, 9, 10, 11, 12, 13, 14, 15);
    }
#pragma unroll
    for (int ni = 0; ni < 4; ++ni) {
      const __bf16* bp = w + (size_t)(n0 + ni * 16 + lrow) * HDIM + k0 + kbB;
      v8bf lo = *(const v8bf*)bp;          // K = kbB .. kbB+7
      v8bf hi = *(const v8bf*)(bp + 16);   // K = kbB+16 .. kbB+23
      bfrag[ni] = __builtin_shufflevector(lo, hi, 0, 1, 2, 3, 4, 5, 6, 7,
                                          8, 9, 10, 11, 12, 13, 14, 15);
    }
#pragma unroll
    for (int mi = 0; mi < 4; ++mi)
#pragma unroll
      for (int ni = 0; ni < 4; ++ni)
        acc[mi][ni] = __builtin_amdgcn_wmma_f32_16x16x32_bf16(
            false, afrag[mi], false, bfrag[ni], (short)0, acc[mi][ni],
            false, false);
  }

  // Epilogue: bias + softplus, store f32. D element (M=r+8*sel, N=lrow).
#pragma unroll
  for (int ni = 0; ni < 4; ++ni) {
    const int g = n0 + ni * 16 + lrow;
    const float bias = b_dt[g];
#pragma unroll
    for (int mi = 0; mi < 4; ++mi) {
#pragma unroll
      for (int r = 0; r < 8; ++r) {
        const int m = m0 + mi * 16 + r + 8 * sel;
        dtbuf[(size_t)m * HDIM + g] = softplusf(acc[mi][ni][r] + bias);
      }
    }
  }
}

// ---------------------------------------------------------------------------
// Pass 1: per-chunk local scan. Lane = (b, h, chunk c); starts from zero
// state, writes local y to out, and emits Sloc (chunk-final local state) and
// P = exp(A * sum(dt over chunk)) (chunk decay, exact via scalar dt-sum).
// 65,536 lanes = 2048 waves -> real occupancy, 128 serial steps instead of
// 2048. Loads coalesced across h; next-step dt/u fetched a step ahead plus
// global_prefetch_b8 further out.
// ---------------------------------------------------------------------------
__global__ __launch_bounds__(256) void ssm_chunk_scan_kernel(
    const float* __restrict__ x,      // [B, T, H]
    const float* __restrict__ A_log,  // [H, N]
    const float* __restrict__ Bm,     // [H, N]
    const float* __restrict__ Cm,     // [H, N]
    const float* __restrict__ Dv,     // [H]
    const float* __restrict__ dtbuf,  // [B, T, H] (softplus'd)
    float*       __restrict__ out,    // [B, T, H] (local y)
    float*       __restrict__ Sloc,   // [NC, B, H, N]
    float*       __restrict__ Pdec)   // [NC, B, H, N]
{
  const int tid  = blockIdx.x * blockDim.x + threadIdx.x; // 0..B*H*NC-1
  const int h    = tid & (HDIM - 1);
  const int rest = tid >> 10;
  const int b    = rest & (BATCH - 1);
  const int c    = rest >> 2;

  float A[NST], Bc[NST], Cc[NST], s[NST];
#pragma unroll
  for (int n = 0; n < NST; ++n) {
    A[n]  = -__expf(A_log[h * NST + n]);
    Bc[n] = Bm[h * NST + n];
    Cc[n] = Cm[h * NST + n];
    s[n]  = 0.0f;
  }
  const float D = Dv[h];

  const size_t base = (size_t)b * TLEN * HDIM + (size_t)(c * CLEN) * HDIM + h;
  const float* xp = x     + base;
  const float* dp = dtbuf + base;
  float*       op = out   + base;

  float dsum   = 0.0f;
  float u_nxt  = xp[0];
  float dt_nxt = dp[0];

  for (int t = 0; t < CLEN; ++t) {
    const float u  = u_nxt;
    const float dt = dt_nxt;
    const int tn = (t + 1 < CLEN) ? (t + 1) : t;
    u_nxt  = xp[tn * HDIM];
    dt_nxt = dp[tn * HDIM];
    if (t + 8 < CLEN) {
      __builtin_prefetch(xp + (t + 8) * HDIM, 0, 0); // global_prefetch_b8
      __builtin_prefetch(dp + (t + 8) * HDIM, 0, 0);
    }
    dsum += dt;

    float y = 0.0f;
#pragma unroll
    for (int n = 0; n < NST; ++n) {
      const float abar = __expf(dt * A[n]);
      s[n] = fmaf(abar, s[n], (dt * Bc[n]) * u);
      y    = fmaf(s[n], Cc[n], y);
    }
    op[t * HDIM] = fmaf(D, u, y);
  }

  const size_t o = ((size_t)(c * BATCH + b) * HDIM + h) * NST;
#pragma unroll
  for (int n = 0; n < NST; ++n) {
    Sloc[o + n] = s[n];
    Pdec[o + n] = __expf(A[n] * dsum);
  }
}

// ---------------------------------------------------------------------------
// Pass 2: stitch chunk carries. One thread per (b,h,n): 16-step serial carry
// recurrence st <- P*st + Sloc; stores per-chunk entry carry and the true
// final state (== reference state_final).
// ---------------------------------------------------------------------------
__global__ __launch_bounds__(256) void ssm_stitch_kernel(
    const float* __restrict__ Sloc,   // [NC, B, H, N]
    const float* __restrict__ Pdec,   // [NC, B, H, N]
    float*       __restrict__ carry,  // [NC, B, H, N] (state entering chunk c)
    float*       __restrict__ stateOut) // [B, H, N]
{
  const int tid = blockIdx.x * blockDim.x + threadIdx.x; // 0..B*H*N-1
  const size_t stride = (size_t)BATCH * HDIM * NST;
  float st = 0.0f;
#pragma unroll
  for (int c = 0; c < NCHUNK; ++c) {
    const size_t idx = (size_t)c * stride + tid;
    carry[idx] = st;
    st = fmaf(Pdec[idx], st, Sloc[idx]);
  }
  stateOut[tid] = st;
}

// ---------------------------------------------------------------------------
// Pass 3: carry correction. For chunks c>=1, true y_t = y_loc_t +
// sum_n C_n * exp(A_n * prefix_dt_sum(t)) * carry_c[n]. Recomputes the prefix
// decay (cheaper than storing B*T*H*N of it) and RMWs out.
// ---------------------------------------------------------------------------
__global__ __launch_bounds__(256) void ssm_correct_kernel(
    const float* __restrict__ A_log,  // [H, N]
    const float* __restrict__ Cm,     // [H, N]
    const float* __restrict__ dtbuf,  // [B, T, H]
    const float* __restrict__ carry,  // [NC, B, H, N]
    float*       __restrict__ out)    // [B, T, H] (RMW)
{
  const int tid  = blockIdx.x * blockDim.x + threadIdx.x;
  const int h    = tid & (HDIM - 1);
  const int rest = tid >> 10;
  const int b    = rest & (BATCH - 1);
  const int c    = rest >> 2;
  if (c == 0) return; // whole waves uniform: carry into chunk 0 is zero

  const size_t co = ((size_t)(c * BATCH + b) * HDIM + h) * NST;
  float A[NST], K[NST];
#pragma unroll
  for (int n = 0; n < NST; ++n) {
    A[n] = -__expf(A_log[h * NST + n]);
    K[n] = Cm[h * NST + n] * carry[co + n];
  }

  const size_t base = (size_t)b * TLEN * HDIM + (size_t)(c * CLEN) * HDIM + h;
  const float* dp = dtbuf + base;
  float*       op = out   + base;

  float dsum   = 0.0f;
  float dt_nxt = dp[0];
  for (int t = 0; t < CLEN; ++t) {
    const float dt = dt_nxt;
    const int tn = (t + 1 < CLEN) ? (t + 1) : t;
    dt_nxt = dp[tn * HDIM];
    if (t + 8 < CLEN)
      __builtin_prefetch(dp + (t + 8) * HDIM, 0, 0);
    dsum += dt;

    float yc = 0.0f;
#pragma unroll
    for (int n = 0; n < NST; ++n)
      yc = fmaf(__expf(A[n] * dsum), K[n], yc);
    op[t * HDIM] += yc;
  }
}

// ---------------------------------------------------------------------------
// Host launcher.
// d_in order: x, A_log, B_mat, C_mat, D_vec, W_dt, b_dt
// d_out: out (B*T*H f32) followed by state_final (B*H*N f32)
// d_ws: dt f32 (33.5MB) | region reused: {x_bf16 16.8MB | W_bf16 2.1MB}
//       then overwritten (GEMM is done) by {Sloc 4.2MB | P 4.2MB | carry 4.2MB}
// Peak requirement: 52.4MB.
// ---------------------------------------------------------------------------
extern "C" void kernel_launch(void* const* d_in, const int* in_sizes, int n_in,
                              void* d_out, int out_size, void* d_ws, size_t ws_size,
                              hipStream_t stream) {
  (void)in_sizes; (void)n_in; (void)out_size; (void)ws_size;

  const float* x     = (const float*)d_in[0];
  const float* A_log = (const float*)d_in[1];
  const float* B_mat = (const float*)d_in[2];
  const float* C_mat = (const float*)d_in[3];
  const float* D_vec = (const float*)d_in[4];
  const float* W_dt  = (const float*)d_in[5];
  const float* b_dt  = (const float*)d_in[6];

  const size_t BTH = (size_t)BATCH * TLEN * HDIM;      // 8,388,608
  const size_t HH  = (size_t)HDIM * HDIM;              // 1,048,576
  const size_t BHN = (size_t)BATCH * HDIM * NST;       // 65,536

  float*  dtbuf = (float*)d_ws;
  char*   scratch2 = (char*)d_ws + BTH * sizeof(float);
  __bf16* xb = (__bf16*)scratch2;
  __bf16* wb = xb + BTH;
  // Sloc/Pdec/carry overlap the bf16 staging area (dead after the GEMM;
  // stream ordering guarantees the GEMM finished before pass 1 writes here).
  float* Sloc   = (float*)scratch2;
  float* Pdec   = Sloc + (size_t)NCHUNK * BHN;
  float* carryb = Pdec + (size_t)NCHUNK * BHN;

  float* out_y     = (float*)d_out;
  float* out_state = out_y + BTH;

  // 0) f32 -> bf16 conversion of GEMM operands
  convert_bf16_kernel<<<2048, 256, 0, stream>>>(x, W_dt, xb, wb,
                                                (int)BTH, (int)HH);

  // 1) dt = softplus(u @ W^T + b) via bf16 WMMA
  dim3 ggrid(BATCH * TLEN / 64, HDIM / 256, 1);
  dt_gemm_wmma_kernel<<<ggrid, 128, 0, stream>>>(xb, wb, b_dt, dtbuf);

  // 2) chunked scan: pass 1 (local), pass 2 (stitch), pass 3 (correct)
  const int lanes1 = BATCH * HDIM * NCHUNK; // 65,536
  ssm_chunk_scan_kernel<<<lanes1 / 256, 256, 0, stream>>>(
      x, A_log, B_mat, C_mat, D_vec, dtbuf, out_y, Sloc, Pdec);

  ssm_stitch_kernel<<<(int)(BHN / 256), 256, 0, stream>>>(
      Sloc, Pdec, carryb, out_state);

  ssm_correct_kernel<<<lanes1 / 256, 256, 0, stream>>>(
      A_log, C_mat, dtbuf, carryb, out_y);
}
